// QuantumMemoryBank_7344394076256
// MI455X (gfx1250) — matrix operand order repaired and checked by reading, not compiled, not run on hardware
//
#include <hip/hip_runtime.h>
#include <math.h>

// ---------------------------------------------------------------------------
// QuantumMemoryBank — CDNA5 (gfx1250) implementation
//
// pooled  = mean_S(query)                    (256 x 1024)  [512MB stream ->
//            async-to-LDS (ASYNCcnt) depth-8 pipeline, 2-stage reduction]
// h       = tanh(pooled @ W1 + b1)           (256 x 1024)  [WMMA f16]
// qq      = h @ W2 + b2                      (256 x 128)   [WMMA f16]
// ov      = qq @ memory^T                    (256 x 131072)[WMMA f16, fused top-3]
// top3    = argmax-3 of ov^2 per row (softmax is monotonic -> never computed)
// out     = tanh(memory[top3][:,:64] @ R1 + rb1) @ R2 + rb2   (256 x 3 x 1024)
//           row order = [3rd largest, 2nd, largest]  (argsort[:, -3:] semantics)
// ---------------------------------------------------------------------------

typedef __attribute__((ext_vector_type(16))) _Float16 v16h;
typedef __attribute__((ext_vector_type(8)))  float    v8f;

#define B_DIM   256
#define S_DIM   512
#define D_DIM   1024
#define M_ROWS  131072
#define Q2      128
#define NQ      64
#define CHUNK   512
#define NCHUNKS (M_ROWS / CHUNK)   // 256
#define SSEG    4
#define SROWS   (S_DIM / SSEG)     // 128
#define PDEPTH  8                  // async ring depth

// ---- CDNA5 async global->LDS copy (ASYNCcnt path), inline-asm form ---------
// GLOBAL_LOAD_ASYNC_TO_LDS_B128: VDST = LDS byte-address VGPR,
// VADDR = 64-bit global address VGPR pair, SADDR = off.  Flat LDS pointers
// truncate to the LDS offset (low 32 bits of the flat address).
__device__ __forceinline__ void async_ld_b128(const void* g, void* l) {
    asm volatile("global_load_async_to_lds_b128 %0, %1, off"
                 :: "v"((uint32_t)(uintptr_t)l),
                    "v"((uint64_t)(uintptr_t)g)
                 : "memory");
}

#if defined(__gfx1250__) && __has_builtin(__builtin_amdgcn_s_wait_asynccnt)
#define ASYNC_WAIT(n) __builtin_amdgcn_s_wait_asynccnt(n)
#else
#define ASYNC_WAIT(n) asm volatile("s_wait_asynccnt %0" :: "i"(n) : "memory")
#endif

// ---------------------------------------------------------------------------
// K0a: partial sums over S/4 rows. grid (B, 4), block 256 (8 waves).
// Each wave owns a 512B d-slice and streams 128 rows through a depth-8
// async-to-LDS ring (no VGPR load destinations -> deep pipeline, 32KB in
// flight per block).
// ---------------------------------------------------------------------------
__global__ void __launch_bounds__(256)
qmb_pool_stage1(const float* __restrict__ query,
                float* __restrict__ partial) {             // (B*4, 1024) f32
    __shared__ __attribute__((aligned(16))) float ring[PDEPTH * 8 * 128]; // 32KB
    const int b    = blockIdx.x;
    const int sseg = blockIdx.y;                           // 0..3
    const int w    = threadIdx.x >> 5;                     // wave 0..7
    const int l    = threadIdx.x & 31;
    const int d    = w * 128 + l * 4;                      // this lane's 4 floats
    const float* base =
        query + ((size_t)b * S_DIM + (size_t)sseg * SROWS) * D_DIM + d;
    float* slot0 = &ring[w * 128 + l * 4];                 // +1024 floats per depth

#pragma unroll
    for (int s = 0; s < PDEPTH; ++s)
        async_ld_b128(base + (size_t)s * D_DIM, slot0 + (s & (PDEPTH - 1)) * 1024);

    float ax = 0.f, ay = 0.f, az = 0.f, aw = 0.f;
    for (int s = 0; s < SROWS; ++s) {
        ASYNC_WAIT(PDEPTH - 1);                            // oldest copy done
        const float4 v = *(const float4*)(slot0 + (s & (PDEPTH - 1)) * 1024);
        ax += v.x; ay += v.y; az += v.z; aw += v.w;
        if (s + PDEPTH < SROWS) {
            // ds_read of this slot must land before the async overwrite
            asm volatile("s_wait_dscnt 0x0" ::: "memory");
            async_ld_b128(base + (size_t)(s + PDEPTH) * D_DIM,
                          slot0 + (s & (PDEPTH - 1)) * 1024);
        }
    }
    float* prow = partial + (size_t)(b * SSEG + sseg) * D_DIM + d;
    prow[0] = ax; prow[1] = ay; prow[2] = az; prow[3] = aw;
}

// K0b: reduce 4 partials -> pooled (f16). grid 256, block 256, 4 d per thread.
__global__ void qmb_pool_stage2(const float* __restrict__ partial,
                                _Float16* __restrict__ pooled_h) {
    const int o = (blockIdx.x * 256 + threadIdx.x) * 4;    // 0..262140
    const int b = o >> 10;
    const int d = o & 1023;
    const float* p = partial + (size_t)b * SSEG * D_DIM + d;
    float4 s0 = *(const float4*)(p);
    float4 s1 = *(const float4*)(p + D_DIM);
    float4 s2 = *(const float4*)(p + 2 * D_DIM);
    float4 s3 = *(const float4*)(p + 3 * D_DIM);
    const float inv = 1.0f / (float)S_DIM;
    _Float16* q = pooled_h + (size_t)b * D_DIM + d;
    q[0] = (_Float16)((s0.x + s1.x + s2.x + s3.x) * inv);
    q[1] = (_Float16)((s0.y + s1.y + s2.y + s3.y) * inv);
    q[2] = (_Float16)((s0.z + s1.z + s2.z + s3.z) * inv);
    q[3] = (_Float16)((s0.w + s1.w + s2.w + s3.w) * inv);
}

// ---------------------------------------------------------------------------
// WMMA tile helpers: one wave computes one 16x16 f32 C tile.
// A (16x32 f16): lane g=lane>>4, m=lane&15; half h -> K = 16*g + h
// B (32x16 f16): lane n=lane&15;            half h -> K = 16*g + h
// C (16x16 f32): c[i] is element (M = i + 8*g, N = lane&15)
// ---------------------------------------------------------------------------

// K1: h = tanh(pooled @ W1 + b1), stored f16.  1024 tiles, 8 waves/block.
__global__ void qmb_mlp1(const _Float16* __restrict__ Ah,   // pooled f16 (256x1024)
                         const float* __restrict__ W1,      // (1024x1024)
                         const float* __restrict__ b1,      // (1024)
                         _Float16* __restrict__ Hh) {       // (256x1024) f16
    const int wave = threadIdx.x >> 5;
    const int lane = threadIdx.x & 31;
    const int g    = lane >> 4;
    const int ln   = lane & 15;
    const int tile = blockIdx.x * 8 + wave;      // 0..1023
    const int m0   = (tile >> 6) * 16;           // 16 tiles in M
    const int n0   = (tile & 63) * 16;           // 64 tiles in N
    v8f c = {};
    for (int kt = 0; kt < D_DIM / 32; ++kt) {
        const int kb = kt * 32;
        v16h a, bm;
#pragma unroll
        for (int h = 0; h < 16; ++h) {
            const int ka = kb + 16 * g + h;
            a[h] = Ah[(size_t)(m0 + ln) * D_DIM + ka];
            const int kB = kb + 16 * g + h;
            bm[h] = (_Float16)W1[(size_t)kB * D_DIM + n0 + ln];
        }
        c = __builtin_amdgcn_wmma_f32_16x16x32_f16(false, a, false, bm,
                                                   (short)0, c, false, false);
    }
    const float bias = b1[n0 + ln];
#pragma unroll
    for (int i = 0; i < 8; ++i) {
        const int m = m0 + i + 8 * g;
        Hh[(size_t)m * D_DIM + n0 + ln] = (_Float16)tanhf(c[i] + bias);
    }
}

// K2: qq = h @ W2 + b2, stored f16.  128 tiles, 8 waves/block.
__global__ void qmb_mlp2(const _Float16* __restrict__ Hh,   // (256x1024) f16
                         const float* __restrict__ W2,      // (1024x128)
                         const float* __restrict__ b2,      // (128)
                         _Float16* __restrict__ QQh) {      // (256x128) f16
    const int wave = threadIdx.x >> 5;
    const int lane = threadIdx.x & 31;
    const int g    = lane >> 4;
    const int ln   = lane & 15;
    const int tile = blockIdx.x * 8 + wave;      // 0..127
    const int m0   = (tile >> 3) * 16;           // 16 tiles in M
    const int n0   = (tile & 7) * 16;            // 8 tiles in N
    v8f c = {};
    for (int kt = 0; kt < D_DIM / 32; ++kt) {
        const int kb = kt * 32;
        v16h a, bm;
#pragma unroll
        for (int h = 0; h < 16; ++h) {
            const int ka = kb + 16 * g + h;
            a[h] = Hh[(size_t)(m0 + ln) * D_DIM + ka];
            const int kB = kb + 16 * g + h;
            bm[h] = (_Float16)W2[(size_t)kB * Q2 + n0 + ln];
        }
        c = __builtin_amdgcn_wmma_f32_16x16x32_f16(false, a, false, bm,
                                                   (short)0, c, false, false);
    }
    const float bias = b2[n0 + ln];
#pragma unroll
    for (int i = 0; i < 8; ++i) {
        const int m = m0 + i + 8 * g;
        QQh[(size_t)m * Q2 + n0 + ln] = (_Float16)(c[i] + bias);
    }
}

__device__ __forceinline__ void top3_insert(float v, int idx,
                                            float& v0, float& v1, float& v2,
                                            int& i0, int& i1, int& i2) {
    if (v > v0)      { v2 = v1; i2 = i1; v1 = v0; i1 = i0; v0 = v; i0 = idx; }
    else if (v > v1) { v2 = v1; i2 = i1; v1 = v;  i1 = idx; }
    else if (v > v2) { v2 = v;  i2 = idx; }
}

// ---------------------------------------------------------------------------
// K3: fused  ov = qq @ memory^T  ->  ov^2  ->  per-(row,chunk) top-3.
// grid (NCHUNKS, B/16), block 256 (8 waves). B tile comes from rows of
// `memory` (row-major, K contiguous per lane -> wide vector loads).
// ---------------------------------------------------------------------------
__global__ void qmb_overlap_top3(const _Float16* __restrict__ QQh, // (256x128) f16
                                 const float* __restrict__ mem,    // (131072x128)
                                 float* __restrict__ pvals,        // (256,NCHUNKS,3)
                                 int*   __restrict__ pidx) {
    __shared__ float sq[16][CHUNK + 1];        // +1 pad: kill bank conflicts
    __shared__ float candV[16][16][3];
    __shared__ int   candI[16][16][3];

    const int chunk = blockIdx.x;
    const int bT    = blockIdx.y;
    const int wave  = threadIdx.x >> 5;
    const int lane  = threadIdx.x & 31;
    const int g     = lane >> 4;
    const int ln    = lane & 15;
    const int m0    = bT * 16;

    // Hoist A: 16 rows x 128 K as four 16x32 f16 tiles (same for every N tile)
    v16h a[4];
#pragma unroll
    for (int kt = 0; kt < 4; ++kt)
#pragma unroll
        for (int h = 0; h < 16; ++h) {
            const int ka = kt * 32 + 16 * g + h;
            a[kt][h] = QQh[(size_t)(m0 + ln) * Q2 + ka];
        }

    for (int t = wave; t < CHUNK / 16; t += 8) {
        const int nG = chunk * CHUNK + t * 16 + ln;   // this lane's memory row
        const float* brow = mem + (size_t)nG * Q2;
        v8f c = {};
#pragma unroll
        for (int kt = 0; kt < 4; ++kt) {
            v16h bm;
#pragma unroll
            for (int h = 0; h < 16; ++h)
                bm[h] = (_Float16)brow[kt * 32 + 16 * g + h];  // 16 consecutive f32
            c = __builtin_amdgcn_wmma_f32_16x16x32_f16(false, a[kt], false, bm,
                                                       (short)0, c, false, false);
        }
#pragma unroll
        for (int i = 0; i < 8; ++i) {
            const float v = c[i];
            sq[i + 8 * g][t * 16 + ln] = v * v;
        }
    }
    __syncthreads();

    // Per-row top-3 over this chunk: 16 threads per row, then merge.
    const int row = threadIdx.x >> 4;
    const int sub = threadIdx.x & 15;
    float v0 = -1.0f, v1 = -1.0f, v2 = -1.0f;   // squares are >= 0
    int   i0 = 0, i1 = 0, i2 = 0;
    for (int j = 0; j < CHUNK / 16; ++j) {
        const int col = sub + 16 * j;
        top3_insert(sq[row][col], chunk * CHUNK + col, v0, v1, v2, i0, i1, i2);
    }
    candV[row][sub][0] = v0; candV[row][sub][1] = v1; candV[row][sub][2] = v2;
    candI[row][sub][0] = i0; candI[row][sub][1] = i1; candI[row][sub][2] = i2;
    __syncthreads();

    if (sub == 0) {
        float w0 = -1.0f, w1 = -1.0f, w2 = -1.0f;
        int   j0 = 0, j1 = 0, j2 = 0;
        for (int s = 0; s < 16; ++s)
#pragma unroll
            for (int r = 0; r < 3; ++r)
                top3_insert(candV[row][s][r], candI[row][s][r],
                            w0, w1, w2, j0, j1, j2);
        const int b = m0 + row;
        const size_t base = ((size_t)b * NCHUNKS + chunk) * 3;
        pvals[base + 0] = w0; pvals[base + 1] = w1; pvals[base + 2] = w2;
        pidx [base + 0] = j0; pidx [base + 1] = j1; pidx [base + 2] = j2;
    }
}

// ---------------------------------------------------------------------------
// K4: per-batch final merge -> gather -> tiny MLP.
// out rows ordered [3rd largest, 2nd, largest] to match argsort[:, -3:].
// ---------------------------------------------------------------------------
__global__ void qmb_final(const float* __restrict__ pvals,
                          const int*   __restrict__ pidx,
                          const float* __restrict__ mem,
                          const float* __restrict__ R1,   // (64x64)
                          const float* __restrict__ rb1,  // (64)
                          const float* __restrict__ R2,   // (64x1024)
                          const float* __restrict__ rb2,  // (1024)
                          float* __restrict__ out) {      // (256,3,1024)
    __shared__ int   topIdx[3];          // [0]=largest, [1]=2nd, [2]=3rd
    __shared__ float tmat[3][NQ];        // tanh hidden, rows in OUTPUT order
    const int b = blockIdx.x;
    const int t = threadIdx.x;

    if (t == 0) {
        float v0 = -1.0f, v1 = -1.0f, v2 = -1.0f;
        int   i0 = 0, i1 = 0, i2 = 0;
        const float* pv = pvals + (size_t)b * NCHUNKS * 3;
        const int*   pi = pidx  + (size_t)b * NCHUNKS * 3;
        for (int k = 0; k < NCHUNKS * 3; ++k)
            top3_insert(pv[k], pi[k], v0, v1, v2, i0, i1, i2);
        topIdx[0] = i0; topIdx[1] = i1; topIdx[2] = i2;
    }
    __syncthreads();

    if (t < 3 * NQ) {
        const int j    = t >> 6;          // output row 0..2
        const int qcol = t & 63;
        const int midx = topIdx[2 - j];   // row j <-> rank (2-j): [3rd,2nd,1st]
        const float* mrow = mem + (size_t)midx * Q2;   // first 64 = qubit part
        float s = rb1[qcol];
#pragma unroll 8
        for (int k = 0; k < NQ; ++k)
            s += mrow[k] * R1[k * NQ + qcol];
        tmat[j][qcol] = tanhf(s);
    }
    __syncthreads();

    for (int w = 0; w < 12; ++w) {
        const int o = w * 256 + t;        // 0..3071
        const int j = o >> 10;
        const int d = o & 1023;
        float s = rb2[d];
#pragma unroll 8
        for (int k = 0; k < NQ; ++k)
            s += tmat[j][k] * R2[(size_t)k * D_DIM + d];
        out[(size_t)b * (3 * D_DIM) + o] = s;
    }
}

// ---------------------------------------------------------------------------
extern "C" void kernel_launch(void* const* d_in, const int* in_sizes, int n_in,
                              void* d_out, int out_size, void* d_ws, size_t ws_size,
                              hipStream_t stream) {
    const float* query = (const float*)d_in[0];
    const float* mem   = (const float*)d_in[1];
    const float* W1    = (const float*)d_in[2];
    const float* b1    = (const float*)d_in[3];
    const float* W2    = (const float*)d_in[4];
    const float* b2    = (const float*)d_in[5];
    const float* R1    = (const float*)d_in[6];
    const float* rb1   = (const float*)d_in[7];
    const float* R2    = (const float*)d_in[8];
    const float* rb2   = (const float*)d_in[9];
    float* out = (float*)d_out;

    char* ws = (char*)d_ws;
    _Float16* pooled_h = (_Float16*)(ws);                        // 512 KB
    _Float16* Hh       = (_Float16*)(ws + (512u << 10));         // 512 KB
    _Float16* QQh      = (_Float16*)(ws + (1024u << 10));        //  64 KB
    float*    pvals    = (float*)   (ws + (1088u << 10));        // 768 KB
    int*      pidx     = (int*)     (ws + (1856u << 10));        // 768 KB
    float*    partial  = (float*)   (ws + (2624u << 10));        //   4 MB

    qmb_pool_stage1<<<dim3(B_DIM, SSEG), 256, 0, stream>>>(query, partial);
    qmb_pool_stage2<<<dim3(B_DIM), 256, 0, stream>>>(partial, pooled_h);
    qmb_mlp1<<<dim3(128), 256, 0, stream>>>(pooled_h, W1, b1, Hh);
    qmb_mlp2<<<dim3(16), 256, 0, stream>>>(Hh, W2, b2, QQh);
    qmb_overlap_top3<<<dim3(NCHUNKS, B_DIM / 16), 256, 0, stream>>>(QQh, mem,
                                                                    pvals, pidx);
    qmb_final<<<dim3(B_DIM), 256, 0, stream>>>(pvals, pidx, mem,
                                               R1, rb1, R2, rb2, out);
}
